// EvenOddConvLayer_28149215658674
// MI455X (gfx1250) — compile-verified, matrix operand order checked by reading, not run on hardware
//
#include <hip/hip_runtime.h>

// ---------------------------------------------------------------------------
// MI455X (gfx1250) EvenOddConvLayer
//   - bf16 WMMA 16x16x32, fp32 accumulate (all K dims are multiples of 32)
//   - one node (20 neighbors -> 32 padded rows -> 2 row-tiles) per wave32
//   - weights pre-transposed to bf16 in d_ws by a prep kernel
//   - branch-free native activations (v_exp/v_log/v_tanh), rolled nt loops to
//     keep VGPR pressure < 256 (no scratch spills)
// ---------------------------------------------------------------------------

typedef __attribute__((ext_vector_type(8)))  __bf16 v8bf;
typedef __attribute__((ext_vector_type(16))) __bf16 v16bf;
typedef __attribute__((ext_vector_type(8)))  float  v8f;

#define NB_M   20     // neighbors per node
#define FE     64     // EVEN
#define FO     64     // ODD
#define FEDGE  32
#define FCAT   224    // 2*64 + 32 + 64

// ws layout (in bf16/ushort elements): transposed weights Wt[n*K + k]
enum : int {
  OFF_EM1 = 0,                 // 64 x 224
  OFF_OGH = OFF_EM1 + 64*224,  // 64 x 224
  OFF_EG  = OFF_OGH + 64*224,  // 64 x 64
  OFF_EM2 = OFF_EG  + 64*64,
  OFF_PEJ = OFF_EM2 + 64*64,
  OFF_PEI = OFF_PEJ + 64*64,
  OFF_OG  = OFF_PEI + 64*64,
  OFF_OM1 = OFF_OG  + 64*64,   // 64 x 128
  WS_ELEMS = OFF_OM1 + 64*128
};

static __device__ __forceinline__ unsigned short f2bf(float f) {
  unsigned u = __builtin_bit_cast(unsigned, f);
  u += 0x7fffu + ((u >> 16) & 1u);           // round to nearest even
  return (unsigned short)(u >> 16);
}
static __device__ __forceinline__ unsigned pack2bf(float a, float b) {
  return (unsigned)f2bf(a) | ((unsigned)f2bf(b) << 16);
}
static __device__ __forceinline__ float bf2f(unsigned short s) {
  unsigned u = ((unsigned)s) << 16;
  return __builtin_bit_cast(float, u);
}

// ---- branch-free fast activations (native v_exp_f32 / v_log_f32 / v_tanh) --
static __device__ __forceinline__ float rcp_fast(float x) {
#if __has_builtin(__builtin_amdgcn_rcpf)
  return __builtin_amdgcn_rcpf(x);
#else
  return 1.f / x;
#endif
}
static __device__ __forceinline__ float tanh_fast(float x) {
#if __has_builtin(__builtin_amdgcn_tanhf)
  return __builtin_amdgcn_tanhf(x);
#elif __has_builtin(__builtin_amdgcn_tanh_f32)
  return __builtin_amdgcn_tanh_f32(x);
#else
  const float t = __expf(-2.f * __builtin_fabsf(x));
  const float r = (1.f - t) * rcp_fast(1.f + t);
  return __builtin_copysignf(r, x);
#endif
}
static __device__ __forceinline__ float softplus_f(float x) {
  // log(1+e^x) = max(x,0) + log(1 + e^-|x|)   (no overflow, no branches)
  return __builtin_fmaxf(x, 0.f) + __logf(1.f + __expf(-__builtin_fabsf(x)));
}
static __device__ __forceinline__ float sigmoid_f(float x) {
  return 0.5f + 0.5f * tanh_fast(0.5f * x);
}

static __device__ __forceinline__ v16bf cat16(v8bf lo, v8bf hi) {
  return __builtin_shufflevector(lo, hi, 0,1,2,3,4,5,6,7,8,9,10,11,12,13,14,15);
}

// A fragment (16x32 bf16) from an LDS tile, row-major, leading dim `ld` elems.
// ISA layout: lanes 0-15 -> row r, K {0..7,16..23}; lanes 16-31 -> K {8..15,24..31}.
static __device__ __forceinline__ v16bf ldsA(const unsigned short* base, int ld,
                                             int koff, int lane) {
  const int r = lane & 15, h = lane >> 4;
  const unsigned short* p = base + r * ld + koff + h * 8;
  v8bf lo = *(const v8bf*)(p);
  v8bf hi = *(const v8bf*)(p + 16);
  return cat16(lo, hi);
}

// B fragment (32x16 bf16) from pre-transposed global weights Wt[n*K + k].
static __device__ __forceinline__ v16bf gloB(const unsigned short* wtb, int K,
                                             int ks, int nt, int lane) {
  const int n = nt * 16 + (lane & 15), h = lane >> 4;
  const unsigned short* p = wtb + n * K + ks * 32 + h * 8;
  v8bf lo = *(const v8bf*)(p);
  v8bf hi = *(const v8bf*)(p + 16);
  return cat16(lo, hi);
}

static __device__ __forceinline__ v8f wmma_bf16(v16bf a, v16bf b, v8f c) {
  return __builtin_amdgcn_wmma_f32_16x16x32_bf16(false, a, false, b,
                                                 (short)0, c, false, false);
}

// ---------------------------------------------------------------------------
// prep: Wt(64 x K, bf16) = transpose(W(K x 64, f32))
// ---------------------------------------------------------------------------
__global__ void wprep_kernel(const float* __restrict__ W,
                             unsigned short* __restrict__ Wt, int K) {
  int idx = blockIdx.x * blockDim.x + threadIdx.x;
  if (idx >= 64 * K) return;
  int n = idx / K, k = idx - n * K;
  Wt[n * K + k] = f2bf(W[k * 64 + n]);
}

// ---------------------------------------------------------------------------
// main kernel: 64 threads = 2 waves, 1 node per wave
// ---------------------------------------------------------------------------
__global__ __launch_bounds__(64)
void eoconv_kernel(const float* __restrict__ even_node,
                   const float* __restrict__ odd_node,
                   const float* __restrict__ nbr_fea,
                   const int*   __restrict__ nbr_idx,
                   const unsigned short* __restrict__ wt,
                   const float* __restrict__ b_em1, const float* __restrict__ b_eg,
                   const float* __restrict__ b_em2, const float* __restrict__ b_pej,
                   const float* __restrict__ b_pei, const float* __restrict__ b_ogh,
                   const float* __restrict__ b_og,
                   float* __restrict__ out_even, float* __restrict__ out_odd,
                   int N) {
  // Per-wave LDS slice (ushort elements):
  //   cat  [32 x 224] @ 0      (7168)
  //   oj   [32 x  64] @ 7168   (2048)   odd_j rows, bf16
  //   hbuf [32 x  64] @ 9216   (2048)   even_h then ogh, bf16
  //   om1A [32 x 128] @ 11264  (4096)   [odd_ie | odd_ei], bf16
  __shared__ __align__(16) unsigned short s_u16[2][15360];
  __shared__ float s_agg[2][128];   // [0:64) even_agg, [64:128) odd_agg

  const int lane = threadIdx.x & 31;
  const int w    = threadIdx.x >> 5;
  const int r16  = lane & 15;
  const int hh   = lane >> 4;

  const int  node_g = blockIdx.x * 2 + w;
  const bool active = node_g < N;
  const int  node   = active ? node_g : (N - 1);

  unsigned short* cat  = &s_u16[w][0];
  unsigned short* oj   = &s_u16[w][7168];
  unsigned short* hbuf = &s_u16[w][9216];
  unsigned short* om1A = &s_u16[w][11264];
  float* aggE = &s_agg[w][0];
  float* aggO = &s_agg[w][64];

  for (int c = lane; c < 128; c += 32) s_agg[w][c] = 0.f;

  // ---- build cat (32 x 224 bf16) and oj (32 x 64 bf16), packed b32 stores --
  {
    const float* ei_row = even_node + (long)node * FE;
    const float* oi_row = odd_node  + (long)node * FO;
    const float2 ei2 = ((const float2*)ei_row)[lane];          // even_i pair
    const float2 oi2 = ((const float2*)oi_row)[lane];          // odd_i  pair
    unsigned* catw = (unsigned*)cat;                           // 112 u32 / row
    unsigned* ojw  = (unsigned*)oj;                            //  32 u32 / row

    for (int r = 0; r < 32; ++r) {
      unsigned* crow = catw + r * (FCAT / 2);
      if (r < NB_M) {
        int idx = nbr_idx[node * NB_M + r];
        idx = idx < 0 ? 0 : (idx >= N ? N - 1 : idx);
        const float2 ej2 = ((const float2*)(even_node + (long)idx * FE))[lane];
        const float2 oj2 = ((const float2*)(odd_node  + (long)idx * FO))[lane];
        crow[lane]      = pack2bf(ei2.x, ei2.y);               // even_i  [0,64)
        crow[32 + lane] = pack2bf(ej2.x, ej2.y);               // even_j  [64,128)
        if (lane < 16) {                                       // nbr_fea [128,160)
          const float2 nb2 =
              ((const float2*)(nbr_fea + ((long)node * NB_M + r) * FEDGE))[lane];
          crow[64 + lane] = pack2bf(nb2.x, nb2.y);
        }
        crow[80 + lane] = pack2bf(oi2.x * oj2.x, oi2.y * oj2.y); // cross [160,224)
        ojw[r * 32 + lane] = pack2bf(oj2.x, oj2.y);
      } else {
        crow[lane] = 0u; crow[32 + lane] = 0u; crow[80 + lane] = 0u;
        if (lane < 16) crow[64 + lane] = 0u;
        ojw[r * 32 + lane] = 0u;
      }
    }
  }
  __syncthreads();

  // ---- per 16-row tile ----
  for (int t = 0; t < 2; ++t) {
    const unsigned short* catT = cat + t * 16 * FCAT;
    const unsigned short* hT   = hbuf + t * 16 * FE;
    const unsigned short* omT  = om1A + t * 16 * 128;

    // (1) even_h = softplus(cat @ W_em1 + b)  -> hbuf (bf16)
#pragma unroll 1
    for (int nt = 0; nt < 4; ++nt) {
      v8f acc = {};
      for (int ks = 0; ks < 7; ++ks)
        acc = wmma_bf16(ldsA(catT, FCAT, ks * 32, lane),
                        gloB(wt + OFF_EM1, FCAT, ks, nt, lane), acc);
      const float bias = b_em1[nt * 16 + r16];
#pragma unroll
      for (int v = 0; v < 8; ++v) {
        const int row = t * 16 + v + 8 * hh;
        hbuf[row * FE + nt * 16 + r16] = f2bf(softplus_f(acc[v] + bias));
      }
    }

    // (2) odd_ie / odd_ei -> om1A (A cols of cat: even_j @ +64, even_i @ +0)
#pragma unroll 1
    for (int nt = 0; nt < 4; ++nt) {
      v8f apj = {}, api = {};
#pragma unroll
      for (int ks = 0; ks < 2; ++ks) {
        apj = wmma_bf16(ldsA(catT + FE, FCAT, ks * 32, lane),
                        gloB(wt + OFF_PEJ, FE, ks, nt, lane), apj);
        api = wmma_bf16(ldsA(catT, FCAT, ks * 32, lane),
                        gloB(wt + OFF_PEI, FE, ks, nt, lane), api);
      }
      const float bpj = b_pej[nt * 16 + r16], bpi = b_pei[nt * 16 + r16];
      const float oi  = odd_node[node * FO + nt * 16 + r16];
#pragma unroll
      for (int v = 0; v < 8; ++v) {
        const int row = t * 16 + v + 8 * hh;
        const float ie = oi * (apj[v] + bpj);
        const float ei = (api[v] + bpi) * bf2f(oj[row * FO + nt * 16 + r16]);
        om1A[row * 128 + nt * 16 + r16]      = f2bf(ie);
        om1A[row * 128 + 64 + nt * 16 + r16] = f2bf(ei);
      }
    }
    __syncthreads();

    // (3) even_gate ⊙ even_msg, masked sum over neighbors (reads hbuf)
#pragma unroll 1
    for (int nt = 0; nt < 4; ++nt) {
      v8f ag = {}, am = {};
#pragma unroll
      for (int ks = 0; ks < 2; ++ks) {
        v16bf a = ldsA(hT, FE, ks * 32, lane);
        ag = wmma_bf16(a, gloB(wt + OFF_EG,  FE, ks, nt, lane), ag);
        am = wmma_bf16(a, gloB(wt + OFF_EM2, FE, ks, nt, lane), am);
      }
      const float bg = b_eg[nt * 16 + r16], bm = b_em2[nt * 16 + r16];
#pragma unroll
      for (int v = 0; v < 8; ++v) {
        const int j = t * 16 + v + 8 * hh;
        if (j < NB_M)
          atomicAdd(&aggE[nt * 16 + r16],
                    sigmoid_f(ag[v] + bg) * softplus_f(am[v] + bm));
      }
    }

    // (4) ogh = softplus(cat @ W_ogh + b) -> hbuf (reuse; step 3 already read it)
#pragma unroll 1
    for (int nt = 0; nt < 4; ++nt) {
      v8f acc = {};
      for (int ks = 0; ks < 7; ++ks)
        acc = wmma_bf16(ldsA(catT, FCAT, ks * 32, lane),
                        gloB(wt + OFF_OGH, FCAT, ks, nt, lane), acc);
      const float bias = b_ogh[nt * 16 + r16];
#pragma unroll
      for (int v = 0; v < 8; ++v) {
        const int row = t * 16 + v + 8 * hh;
        hbuf[row * FE + nt * 16 + r16] = f2bf(softplus_f(acc[v] + bias));
      }
    }
    __syncthreads();

    // (5) fused: odd_value = tanh(om1A @ W_om1); odd_gate = sigmoid(ogh @ W_og + b)
    //     aggO += gate ⊙ value  (no val[] array -> no dynamic reg indexing)
#pragma unroll 1
    for (int nt = 0; nt < 4; ++nt) {
      v8f av = {};
      for (int ks = 0; ks < 4; ++ks)
        av = wmma_bf16(ldsA(omT, 128, ks * 32, lane),
                       gloB(wt + OFF_OM1, 128, ks, nt, lane), av);
      v8f ac = {};
#pragma unroll
      for (int ks = 0; ks < 2; ++ks)
        ac = wmma_bf16(ldsA(hT, FE, ks * 32, lane),
                       gloB(wt + OFF_OG, FE, ks, nt, lane), ac);
      const float bo = b_og[nt * 16 + r16];
#pragma unroll
      for (int v = 0; v < 8; ++v) {
        const int j = t * 16 + v + 8 * hh;
        if (j < NB_M)
          atomicAdd(&aggO[nt * 16 + r16],
                    sigmoid_f(ac[v] + bo) * tanh_fast(av[v]));
      }
    }
    __syncthreads();   // hbuf/om1A rewritten next t iteration
  }
  __syncthreads();

  if (active) {
#pragma unroll
    for (int c0 = 0; c0 < 2; ++c0) {
      const int c = lane + 32 * c0;
      out_even[node * FE + c] = even_node[node * FE + c] + aggE[c];
      out_odd [node * FO + c] = odd_node [node * FO + c] + aggO[c];
    }
  }
}

// ---------------------------------------------------------------------------
extern "C" void kernel_launch(void* const* d_in, const int* in_sizes, int n_in,
                              void* d_out, int out_size, void* d_ws, size_t ws_size,
                              hipStream_t stream) {
  const float* even_node = (const float*)d_in[0];
  const float* odd_node  = (const float*)d_in[1];
  const float* nbr_fea   = (const float*)d_in[2];
  const int*   nbr_idx   = (const int*)  d_in[3];
  const float* W_em1 = (const float*)d_in[4];
  const float* b_em1 = (const float*)d_in[5];
  const float* W_eg  = (const float*)d_in[6];
  const float* b_eg  = (const float*)d_in[7];
  const float* W_em2 = (const float*)d_in[8];
  const float* b_em2 = (const float*)d_in[9];
  const float* W_pej = (const float*)d_in[10];
  const float* b_pej = (const float*)d_in[11];
  const float* W_pei = (const float*)d_in[12];
  const float* b_pei = (const float*)d_in[13];
  const float* W_om1 = (const float*)d_in[14];
  const float* W_ogh = (const float*)d_in[15];
  const float* b_ogh = (const float*)d_in[16];
  const float* W_og  = (const float*)d_in[17];
  const float* b_og  = (const float*)d_in[18];

  unsigned short* wt = (unsigned short*)d_ws;

  auto prep = [&](const float* W, int off, int K) {
    const int tot = 64 * K;
    wprep_kernel<<<(tot + 255) / 256, 256, 0, stream>>>(W, wt + off, K);
  };
  prep(W_em1, OFF_EM1, 224);
  prep(W_ogh, OFF_OGH, 224);
  prep(W_eg,  OFF_EG,  64);
  prep(W_em2, OFF_EM2, 64);
  prep(W_pej, OFF_PEJ, 64);
  prep(W_pei, OFF_PEI, 64);
  prep(W_og,  OFF_OG,  64);
  prep(W_om1, OFF_OM1, 128);

  const int N = in_sizes[0] / FE;
  float* out_even = (float*)d_out;
  float* out_odd  = out_even + (long)N * FE;

  const int blocks = (N + 1) / 2;   // 2 waves per block, 1 node per wave
  eoconv_kernel<<<blocks, 64, 0, stream>>>(
      even_node, odd_node, nbr_fea, nbr_idx, wt,
      b_em1, b_eg, b_em2, b_pej, b_pei, b_ogh, b_og,
      out_even, out_odd, N);
}
